// GATLayer_55293408969146
// MI455X (gfx1250) — matrix operand order repaired and checked by reading, not compiled, not run on hardware
//
#include <hip/hip_runtime.h>
#include <math.h>

// GAT layer, MI455X (gfx1250, wave32).
// N=8192 nodes, IN_F=512, OUT_F=64, ALPHA=0.2.
// Memory-bound on adj (268 MB, read exactly once, nontemporal b128 stream).
// x@W in exact f32 WMMA (feeds exp-sensitive logits); attn@Wh in f16 WMMA
// (K=32) with f32 accumulation.

#define N_NODES 8192
#define IN_F    512
#define OUT_F   64
#define ALPHA   0.2f

typedef __attribute__((ext_vector_type(2)))  float    v2f;
typedef __attribute__((ext_vector_type(4)))  float    v4f;
typedef __attribute__((ext_vector_type(8)))  float    v8f;
typedef __attribute__((ext_vector_type(4)))  int      v4i;
typedef __attribute__((ext_vector_type(8)))  _Float16 v8h;
typedef __attribute__((ext_vector_type(16))) _Float16 v16h;

static __device__ __forceinline__ v8f wmma_f32_4(v2f a, v2f b, v8f c) {
  // D = A(16x4) * B(4x16) + C(16x16), fp32
  return __builtin_amdgcn_wmma_f32_16x16x4_f32(
      false, a, false, b, (short)0, c, false, false);
}

static __device__ __forceinline__ v8f wmma_f16_32(v16h a, v16h b, v8f c) {
  // D = A(16x32) * B(32x16) + C(16x16), f16 in / f32 accumulate
  return __builtin_amdgcn_wmma_f32_16x16x32_f16(
      false, a, false, b, (short)0, c, false, false);
}

// ---------------------------------------------------------------------------
// Kernel 1: Wh = x @ W (8192x512 @ 512x64), one wave per 16-row tile.
// Emits Wh in f32 (row-major, for the f/g dot products) AND WhT16 in f16
// (transposed 64 x 8192, for b128 B-fragment loads in the attention kernel).
// ---------------------------------------------------------------------------
__global__ __launch_bounds__(32) void k_gemm_xW(const float* __restrict__ x,
                                                const float* __restrict__ W,
                                                float* __restrict__ Wh,
                                                _Float16* __restrict__ WhT16) {
  const int lane = threadIdx.x & 31;
  const int hi   = lane >> 4;
  const int r16  = lane & 15;
  const int m0   = blockIdx.x * 16;

  v8f z = (v8f)0.0f;
  v8f acc[4] = {z, z, z, z};

  const float* xrow = x + (size_t)(m0 + r16) * IN_F;

  for (int k = 0; k < IN_F; k += 4) {
    v2f a = *(const v2f*)(xrow + k + 2 * hi);            // 8B-aligned b64
    const float* w0 = W + (size_t)(k + 2 * hi) * OUT_F + r16;
#pragma unroll
    for (int n = 0; n < 4; ++n) {
      v2f b = { w0[n * 16], w0[OUT_F + n * 16] };
      acc[n] = wmma_f32_4(a, b, acc[n]);
    }
  }

#pragma unroll
  for (int n = 0; n < 4; ++n) {
    // f32 row-major store (C layout: VGPR r -> row r + 8*hi, col n*16+r16)
#pragma unroll
    for (int r = 0; r < 8; ++r)
      Wh[(size_t)(m0 + r + 8 * hi) * OUT_F + n * 16 + r16] = acc[n][r];
    // f16 transposed store: fixed col c = n*16+r16, 8 consecutive rows -> b128
    v8h h;
#pragma unroll
    for (int r = 0; r < 8; ++r) h[r] = (_Float16)acc[n][r];
    *(v8h*)(WhT16 + (size_t)(n * 16 + r16) * N_NODES + m0 + 8 * hi) = h;
  }
}

// ---------------------------------------------------------------------------
// Kernel 2: f[i] = Wh[i].a1 , g[i] = Wh[i].a2  (exact f32)
// ---------------------------------------------------------------------------
__global__ __launch_bounds__(256) void k_fg(const float* __restrict__ Wh,
                                            const float* __restrict__ a,
                                            float* __restrict__ f,
                                            float* __restrict__ g) {
  const int i = blockIdx.x * 256 + threadIdx.x;
  const float* wr = Wh + (size_t)i * OUT_F;
  float s1 = 0.f, s2 = 0.f;
#pragma unroll 8
  for (int c = 0; c < OUT_F; ++c) {
    const float w = wr[c];
    s1 = fmaf(w, a[c], s1);
    s2 = fmaf(w, a[OUT_F + c], s2);
  }
  f[i] = s1;
  g[i] = s2;
}

// ---------------------------------------------------------------------------
// Kernel 2b: gmax = max_j g[j]  (single block reduction)
// ---------------------------------------------------------------------------
__global__ __launch_bounds__(1024) void k_gmax(const float* __restrict__ g,
                                               float* __restrict__ gmax) {
  __shared__ float s[1024];
  float m = -3.4e38f;
  for (int i = threadIdx.x; i < N_NODES; i += 1024) m = fmaxf(m, g[i]);
  s[threadIdx.x] = m;
  __syncthreads();
  for (int off = 512; off > 0; off >>= 1) {
    if (threadIdx.x < off) s[threadIdx.x] = fmaxf(s[threadIdx.x], s[threadIdx.x + off]);
    __syncthreads();
  }
  if (threadIdx.x == 0) *gmax = s[0];
}

// ---------------------------------------------------------------------------
// Kernel 3: fused masked softmax + attn@Wh, single pass over adj.
// Row max bound M_i = leakyrelu(f_i + gmax): leakyrelu is monotonic, so this
// is the exact unmasked row max; masked max <= M_i => exp(e - M_i) <= 1.
// One block = 16 output rows, 4 waves split the 256 32-column j-tiles.
// p is produced directly in the f16 A-fragment layout: lane L holds columns
// j0 + 8*hi + 0..7 and j0 + 16 + 8*hi + 0..7 (two contiguous runs -> b128
// loads of adj and g).
// ---------------------------------------------------------------------------
__global__ __launch_bounds__(128) void k_attn(const int* __restrict__ adj,
                                              const _Float16* __restrict__ WhT16,
                                              const float* __restrict__ f,
                                              const float* __restrict__ g,
                                              const float* __restrict__ gmaxp,
                                              float* __restrict__ out) {
  __shared__ float s_acc[4][16][OUT_F];   // 16 KB partial accumulators
  __shared__ float s_l[4][16];            // partial row sums

  const int tid  = threadIdx.x;
  const int wave = tid >> 5;
  const int lane = tid & 31;
  const int hi   = lane >> 4;
  const int r16  = lane & 15;
  const int m0   = blockIdx.x * 16;
  const int row  = m0 + r16;

  const float fi = f[row];
  float M = fi + *gmaxp;
  M = M > 0.f ? M : ALPHA * M;

  v8f z = (v8f)0.0f;
  v8f acc[4] = {z, z, z, z};
  float l = 0.f;

  const size_t arow = (size_t)row * N_NODES;

  for (int jt = wave; jt < N_NODES / 32; jt += 4) {
    const int j0 = jt * 32;

    // Build the 16x32 f16 A fragment (p values) + accumulate row sum in f32.
    v16h A;
#pragma unroll
    for (int t = 0; t < 2; ++t) {
      const int jb = j0 + 16 * t + 8 * hi;           // 32B-aligned
      const v4i a0 = __builtin_nontemporal_load((const v4i*)(adj + arow + jb));
      const v4i a1 = __builtin_nontemporal_load((const v4i*)(adj + arow + jb + 4));
      const v4f g0 = *(const v4f*)(g + jb);
      const v4f g1 = *(const v4f*)(g + jb + 4);
#pragma unroll
      for (int k = 0; k < 4; ++k) {
        float e0 = fi + g0[k];  e0 = e0 > 0.f ? e0 : ALPHA * e0;
        const float p0 = a0[k] ? __expf(e0 - M) : 0.f;
        float e1 = fi + g1[k];  e1 = e1 > 0.f ? e1 : ALPHA * e1;
        const float p1 = a1[k] ? __expf(e1 - M) : 0.f;
        l += p0 + p1;
        A[t * 8 + k]     = (_Float16)p0;
        A[t * 8 + 4 + k] = (_Float16)p1;
      }
    }

    // acc[n] += p(16x32) @ Wh[j0:j0+32, n*16:(n+1)*16]
    // B fragment: halves k = 8*hi + 0..7 and 16 + 8*hi + 0..7 of transposed Wh -> two b128 loads.
#pragma unroll
    for (int n = 0; n < 4; ++n) {
      const _Float16* wp = WhT16 + (size_t)(n * 16 + r16) * N_NODES + j0 + 8 * hi;
      const v8h b0 = *(const v8h*)(wp);
      const v8h b1 = *(const v8h*)(wp + 16);
      const v16h B = __builtin_shufflevector(b0, b1, 0, 1, 2, 3, 4, 5, 6, 7,
                                             8, 9, 10, 11, 12, 13, 14, 15);
      acc[n] = wmma_f16_32(A, B, acc[n]);
    }
  }

  // full row sum: combine lane pair (lane, lane^16)
  l += __shfl_xor(l, 16, 32);

  // dump wave partials to LDS (C layout: VGPR r -> row r + 8*hi)
#pragma unroll
  for (int n = 0; n < 4; ++n)
#pragma unroll
    for (int r = 0; r < 8; ++r)
      s_acc[wave][r + 8 * hi][n * 16 + r16] = acc[n][r];
  if (hi == 0) s_l[wave][r16] = l;
  __syncthreads();

  // 128 threads reduce the 16x64 block and normalize
  const int orow = tid >> 3;        // 0..15
  const int c0   = (tid & 7) * 8;   // 0..56
  const float lsum = s_l[0][orow] + s_l[1][orow] + s_l[2][orow] + s_l[3][orow];
  const float inv = 1.0f / lsum;
#pragma unroll
  for (int c = 0; c < 8; ++c) {
    const float v = s_acc[0][orow][c0 + c] + s_acc[1][orow][c0 + c] +
                    s_acc[2][orow][c0 + c] + s_acc[3][orow][c0 + c];
    out[(size_t)(m0 + orow) * OUT_F + c0 + c] = v * inv;
  }
}

// ---------------------------------------------------------------------------
extern "C" void kernel_launch(void* const* d_in, const int* in_sizes, int n_in,
                              void* d_out, int out_size, void* d_ws, size_t ws_size,
                              hipStream_t stream) {
  const float* x   = (const float*)d_in[0];
  const int*   adj = (const int*)d_in[1];
  const float* W   = (const float*)d_in[2];
  const float* a   = (const float*)d_in[3];
  float* out = (float*)d_out;

  // workspace: WhT16 f16 (64x8192, 1MB) | Wh f32 (8192x64, 2MB) | f | g | gmax
  _Float16* WhT16 = (_Float16*)d_ws;
  float* Wh   = (float*)((char*)d_ws + (size_t)OUT_F * N_NODES * sizeof(_Float16));
  float* f    = Wh + (size_t)N_NODES * OUT_F;
  float* g    = f + N_NODES;
  float* gmax = g + N_NODES;

  k_gemm_xW<<<N_NODES / 16, 32, 0, stream>>>(x, W, Wh, WhT16);
  k_fg<<<N_NODES / 256, 256, 0, stream>>>(Wh, a, f, g);
  k_gmax<<<1, 1024, 0, stream>>>(g, gmax);
  k_attn<<<N_NODES / 16, 128, 0, stream>>>(adj, WhT16, f, g, gmax, out);
}
